// MambaBlock_8615704395885
// MI455X (gfx1250) — matrix operand order repaired and checked by reading, not compiled, not run on hardware
//
#include <hip/hip_runtime.h>
#include <hip/hip_bf16.h>
#include <math.h>

// ---------------------------------------------------------------------------
// Problem constants: B=4, L=16384, D_MODEL=256, D_INNER=512, P=64. M = 65536.
// ---------------------------------------------------------------------------
#define BB      4
#define LLEN    16384
#define DMODEL  256
#define DINNER  512
#define PP      64
#define MROWS   (BB * LLEN)          // 65536

typedef __attribute__((ext_vector_type(16))) __bf16 v16bf;
typedef __attribute__((ext_vector_type(8)))  float  v8f;

static __device__ inline v8f vzero8() {
    v8f z;
#pragma unroll
    for (int i = 0; i < 8; ++i) z[i] = 0.0f;
    return z;
}

// fast silu: v * rcp(1+exp(-v))  (v_rcp_f32 instead of IEEE divide)
static __device__ inline float silu_f(float v) {
    return v * __builtin_amdgcn_rcpf(1.0f + __expf(-v));
}

// CDNA5 async copy global->LDS (ASYNCcnt-tracked), GV mode, 16B per lane.
static __device__ inline void async_ld_b128(unsigned lds_byte_off, const void* gaddr) {
    asm volatile("global_load_async_to_lds_b128 %0, %1, off"
                 :: "v"(lds_byte_off), "v"(gaddr) : "memory");
}
static __device__ inline void wait_asynccnt0() {
    asm volatile("s_wait_asynccnt 0x0" ::: "memory");
}

// ---------------------------------------------------------------------------
// Kernel 0: parameter prep (discretization + bf16 N-major weight layouts)
// ---------------------------------------------------------------------------
__global__ __launch_bounds__(256) void k_setup(
    const float* __restrict__ W_in, const float* __restrict__ W_out,
    const float* __restrict__ A_real, const float* __restrict__ A_imag,
    const float* __restrict__ B_re, const float* __restrict__ B_im,
    const float* __restrict__ C_re, const float* __restrict__ C_im,
    const float* __restrict__ inv_dt,
    __bf16* __restrict__ Wt, __bf16* __restrict__ Wo,
    __bf16* __restrict__ dBb, __bf16* __restrict__ Cbm,
    float* __restrict__ dAc)
{
    const long idx = (long)blockIdx.x * 256 + threadIdx.x;   // grid: 1024 blocks

    if (idx < 1024L * 256) {               // Wt[n][k] = W_in[k][n]
        long n = idx >> 8, k = idx & 255;
        Wt[n * 256 + k] = (__bf16)W_in[k * 1024 + n];
    }
    if (idx < 256L * 512) {                // Wo[n][k] = W_out[k][n]
        long n = idx >> 9, k = idx & 511;
        Wo[n * 512 + k] = (__bf16)W_out[k * 256 + n];
    }
    if (idx < 128L * 512) {                // dBb rows 0..63 re, 64..127 im
        long n = idx >> 9, k = idx & 511;
        int p = (int)(n & 63);
        float dt = log1pf(__expf(inv_dt[p]));
        float Ar = A_real[p], Ai = A_imag[p];
        float xr = 1.0f - 0.5f * dt * Ar, xi = -0.5f * dt * Ai;
        float den = xr * xr + xi * xi;
        float blr = xr / den, bli = -xi / den;
        float sr = blr * dt, si = bli * dt;
        float br = B_re[p * 512 + k], bi = B_im[p * 512 + k];
        float v = (n < 64) ? (sr * br - si * bi) : (sr * bi + si * br);
        dBb[n * 512 + k] = (__bf16)v;
    }
    if (idx < 512L * 128) {                // Cbm[j][k] = k<64 ? C_re : -C_im
        long j = idx >> 7, k = idx & 127;
        float v = (k < 64) ? C_re[j * 64 + k] : -C_im[j * 64 + (k - 64)];
        Cbm[j * 128 + k] = (__bf16)v;
    }
    if (idx < PP) {                        // dA = BL*(1 + dt/2 A)
        int p = (int)idx;
        float dt = log1pf(__expf(inv_dt[p]));
        float Ar = A_real[p], Ai = A_imag[p];
        float xr = 1.0f - 0.5f * dt * Ar, xi = -0.5f * dt * Ai;
        float den = xr * xr + xi * xi;
        float blr = xr / den, bli = -xi / den;
        float yr = 1.0f + 0.5f * dt * Ar, yi = 0.5f * dt * Ai;
        dAc[2 * p]     = blr * yr - bli * yi;
        dAc[2 * p + 1] = blr * yi + bli * yr;
    }
}

// ---------------------------------------------------------------------------
// Kernel 1: 32-row tile -> in_proj (WMMA, N=1024, B-frag reused across 2
// row-subtiles), SiLU, then Bu = u @ dB^T (WMMA, K=512, N=128 re|im packed).
// 8 waves / block, wave32.
// ---------------------------------------------------------------------------
__global__ __launch_bounds__(256) void k_inproj_bu(
    const float* __restrict__ x,        // (M, 256)
    const __bf16* __restrict__ Wt,      // (1024, 256) N-major
    const __bf16* __restrict__ dBb,     // (128, 512)  N-major
    __bf16* __restrict__ u_out,         // (M, 512)
    __bf16* __restrict__ g_out,         // (M, 512)
    float* __restrict__ Bu)             // (M, 128)
{
    __shared__ __bf16 xl[32 * DMODEL];  // 16 KB
    __shared__ __bf16 ul[32 * DINNER];  // 32 KB

    const int tid   = threadIdx.x;
    const int lane  = tid & 31;
    const int wave  = tid >> 5;
    const int row16 = lane & 15;
    const int half  = lane >> 4;
    const long m0   = (long)blockIdx.x * 32;

    // stage x tile as bf16 (row-major [32][256])
    for (int i = tid; i < 32 * DMODEL; i += 256) {
        int r = i >> 8, c = i & 255;
        xl[i] = (__bf16)x[(m0 + r) * DMODEL + c];
    }
    __syncthreads();

    // ---- in_proj: wave covers 128 N-cols of 1024, 2 row-subtiles ----------
    v8f acc[2][8];
#pragma unroll
    for (int s = 0; s < 2; ++s)
#pragma unroll
        for (int t = 0; t < 8; ++t) acc[s][t] = vzero8();

    for (int kb = 0; kb < DMODEL; kb += 32) {
        v16bf a[2];
#pragma unroll
        for (int s = 0; s < 2; ++s) {
            const __bf16* pa = &xl[(s * 16 + row16) * DMODEL + kb + 8 * half];
#pragma unroll
            for (int i = 0; i < 8; ++i) { a[s][i] = pa[i]; a[s][8 + i] = pa[16 + i]; }
        }
#pragma unroll
        for (int t = 0; t < 8; ++t) {
            int n = wave * 128 + t * 16 + row16;
            const __bf16* pb = &Wt[(long)n * 256 + kb + 8 * half];
            v16bf bf;
#pragma unroll
            for (int i = 0; i < 8; ++i) { bf[i] = pb[i]; bf[8 + i] = pb[16 + i]; }
            acc[0][t] = __builtin_amdgcn_wmma_f32_16x16x32_bf16(
                false, a[0], false, bf, (short)0, acc[0][t], false, false);
            acc[1][t] = __builtin_amdgcn_wmma_f32_16x16x32_bf16(
                false, a[1], false, bf, (short)0, acc[1][t], false, false);
        }
    }

    // SiLU + scatter: cols <512 -> u (LDS + global), else gate g (global)
#pragma unroll
    for (int s = 0; s < 2; ++s) {
#pragma unroll
        for (int t = 0; t < 8; ++t) {
            int cg = wave * 128 + t * 16 + row16;
#pragma unroll
            for (int j = 0; j < 8; ++j) {
                int lr = s * 16 + j + 8 * half;
                float v = silu_f(acc[s][t][j]);
                if (cg < DINNER) {
                    ul[lr * DINNER + cg] = (__bf16)v;
                    u_out[(m0 + lr) * DINNER + cg] = (__bf16)v;
                } else {
                    g_out[(m0 + lr) * DINNER + (cg - DINNER)] = (__bf16)v;
                }
            }
        }
    }
    __syncthreads();

    // ---- Bu GEMM: (32 x 512) @ (512 x 128), wave = one 16-col strip -------
    v8f bacc[2];
#pragma unroll
    for (int s = 0; s < 2; ++s) bacc[s] = vzero8();
    const int n0 = wave * 16;
    for (int kb = 0; kb < DINNER; kb += 32) {
        v16bf a[2];
#pragma unroll
        for (int s = 0; s < 2; ++s) {
            const __bf16* pa = &ul[(s * 16 + row16) * DINNER + kb + 8 * half];
#pragma unroll
            for (int i = 0; i < 8; ++i) { a[s][i] = pa[i]; a[s][8 + i] = pa[16 + i]; }
        }
        const __bf16* pb = &dBb[(long)(n0 + row16) * 512 + kb + 8 * half];
        v16bf bf;
#pragma unroll
        for (int i = 0; i < 8; ++i) { bf[i] = pb[i]; bf[8 + i] = pb[16 + i]; }
        bacc[0] = __builtin_amdgcn_wmma_f32_16x16x32_bf16(
            false, a[0], false, bf, (short)0, bacc[0], false, false);
        bacc[1] = __builtin_amdgcn_wmma_f32_16x16x32_bf16(
            false, a[1], false, bf, (short)0, bacc[1], false, false);
    }
#pragma unroll
    for (int s = 0; s < 2; ++s)
#pragma unroll
        for (int j = 0; j < 8; ++j) {
            int lr = s * 16 + j + 8 * half;
            Bu[(m0 + lr) * 128 + n0 + row16] = bacc[s][j];
        }
}

// ---------------------------------------------------------------------------
// Kernel 2: complex scan h_t = dA*h_{t-1} + Bu_t per (b,p).
// 256 threads x 64-step segments + Hillis-Steele over aggregates in LDS.
// ---------------------------------------------------------------------------
__global__ __launch_bounds__(256) void k_scan(
    const float* __restrict__ dAc,
    const float* __restrict__ Bu,       // (M, 128)
    float* __restrict__ h)              // (M, 128)
{
    __shared__ float sar[256], sai[256], sbr[256], sbi[256];

    const int b = blockIdx.x >> 6;
    const int p = blockIdx.x & 63;
    const float dr = dAc[2 * p], di = dAc[2 * p + 1];
    const int tid = threadIdx.x;
    const long base = ((long)b * LLEN + (long)tid * 64) * 128;

    float ar = 1.0f, ai = 0.0f, hr = 0.0f, hi = 0.0f;
    for (int i = 0; i < 64; ++i) {
        float br = Bu[base + (long)i * 128 + p];
        float bi = Bu[base + (long)i * 128 + 64 + p];
        float nr = dr * hr - di * hi + br;
        float ni = dr * hi + di * hr + bi;
        hr = nr; hi = ni;
        float tr = dr * ar - di * ai;
        float ti = dr * ai + di * ar;
        ar = tr; ai = ti;
    }
    sar[tid] = ar; sai[tid] = ai; sbr[tid] = hr; sbi[tid] = hi;
    __syncthreads();

    for (int off = 1; off < 256; off <<= 1) {
        float par = 0, pai = 0, pbr = 0, pbi = 0;
        const bool act = (tid >= off);
        if (act) { par = sar[tid - off]; pai = sai[tid - off];
                   pbr = sbr[tid - off]; pbi = sbi[tid - off]; }
        __syncthreads();
        if (act) {
            float car = sar[tid], cai = sai[tid];
            float cbr = sbr[tid], cbi = sbi[tid];
            sar[tid] = car * par - cai * pai;
            sai[tid] = car * pai + cai * par;
            sbr[tid] = car * pbr - cai * pbi + cbr;
            sbi[tid] = car * pbi + cai * pbr + cbi;
        }
        __syncthreads();
    }

    float pr = 0.0f, pi = 0.0f;
    if (tid > 0) { pr = sbr[tid - 1]; pi = sbi[tid - 1]; }

    hr = pr; hi = pi;
    for (int i = 0; i < 64; ++i) {
        float br = Bu[base + (long)i * 128 + p];
        float bi = Bu[base + (long)i * 128 + 64 + p];
        float nr = dr * hr - di * hi + br;
        float ni = dr * hi + di * hr + bi;
        hr = nr; hi = ni;
        h[base + (long)i * 128 + p]      = hr;
        h[base + (long)i * 128 + 64 + p] = hi;
    }
}

// ---------------------------------------------------------------------------
// Kernel 3: y = (Re(h C^T) + D*u) * g ; out = y @ W_out  (both WMMA).
// u/g tiles staged via CDNA5 async global->LDS copies (ASYNCcnt).
// ---------------------------------------------------------------------------
__global__ __launch_bounds__(256) void k_out(
    const float* __restrict__ h,        // (M, 128)
    const __bf16* __restrict__ Cbm,     // (512, 128) N-major
    const __bf16* __restrict__ u_b,     // (M, 512)
    const __bf16* __restrict__ g_b,     // (M, 512)
    const float* __restrict__ Dv,       // (512)
    const __bf16* __restrict__ Wo,      // (256, 512) N-major
    float* __restrict__ out)            // (M, 256)
{
    __shared__ __bf16 hl[16 * 128];     // 4 KB
    __shared__ __bf16 yl[16 * DINNER];  // 16 KB
    __shared__ __bf16 ulu[16 * DINNER]; // 16 KB (async-staged u tile)
    __shared__ __bf16 glg[16 * DINNER]; // 16 KB (async-staged g tile)

    const int tid   = threadIdx.x;
    const int lane  = tid & 31;
    const int wave  = tid >> 5;
    const int row16 = lane & 15;
    const int half  = lane >> 4;
    const long m0   = (long)blockIdx.x * 16;

    // async-stage u and g tiles (16 x 512 bf16 = 1024 x 16B chunks each)
    {
        const unsigned ubase = (unsigned)(size_t)(&ulu[0]);
        const unsigned gbase = (unsigned)(size_t)(&glg[0]);
        for (int c = tid; c < 1024; c += 256) {
            int r = c >> 6, col = (c & 63) << 3;       // 8 bf16 = 16B per chunk
            long gi = (m0 + r) * DINNER + col;
            async_ld_b128(ubase + (unsigned)c * 16u, &u_b[gi]);
            async_ld_b128(gbase + (unsigned)c * 16u, &g_b[gi]);
        }
    }

    // stage h tile as bf16 (VALU path, overlaps with async copies)
    for (int i = tid; i < 16 * 128; i += 256) {
        int r = i >> 7, c = i & 127;
        hl[i] = (__bf16)h[(m0 + r) * 128 + c];
    }
    wait_asynccnt0();
    __syncthreads();

    // ---- y_inner = h @ Cbm^T : wave covers 64 N-cols of 512 ---------------
    v8f acc[4];
#pragma unroll
    for (int t = 0; t < 4; ++t) acc[t] = vzero8();

    for (int kb = 0; kb < 128; kb += 32) {
        v16bf a;
        const __bf16* pa = &hl[row16 * 128 + kb + 8 * half];
#pragma unroll
        for (int i = 0; i < 8; ++i) { a[i] = pa[i]; a[8 + i] = pa[16 + i]; }
#pragma unroll
        for (int t = 0; t < 4; ++t) {
            int n = wave * 64 + t * 16 + row16;
            const __bf16* pb = &Cbm[(long)n * 128 + kb + 8 * half];
            v16bf bf;
#pragma unroll
            for (int i = 0; i < 8; ++i) { bf[i] = pb[i]; bf[8 + i] = pb[16 + i]; }
            acc[t] = __builtin_amdgcn_wmma_f32_16x16x32_bf16(
                false, a, false, bf, (short)0, acc[t], false, false);
        }
    }

    // elementwise: + D*u, * g  (u/g from LDS), stage y as bf16
#pragma unroll
    for (int t = 0; t < 4; ++t) {
        int cg = wave * 64 + t * 16 + row16;
        float Dj = Dv[cg];
#pragma unroll
        for (int j = 0; j < 8; ++j) {
            int lr = j + 8 * half;
            float uu = (float)ulu[lr * DINNER + cg];
            float gg = (float)glg[lr * DINNER + cg];
            float yv = (acc[t][j] + Dj * uu) * gg;
            yl[lr * DINNER + cg] = (__bf16)yv;
        }
    }
    __syncthreads();

    // ---- out = y @ W_out^T : wave covers 32 N-cols of 256 -----------------
    v8f oacc[2];
#pragma unroll
    for (int t = 0; t < 2; ++t) oacc[t] = vzero8();

    for (int kb = 0; kb < DINNER; kb += 32) {
        v16bf a;
        const __bf16* pa = &yl[row16 * DINNER + kb + 8 * half];
#pragma unroll
        for (int i = 0; i < 8; ++i) { a[i] = pa[i]; a[8 + i] = pa[16 + i]; }
#pragma unroll
        for (int t = 0; t < 2; ++t) {
            int n = wave * 32 + t * 16 + row16;
            const __bf16* pb = &Wo[(long)n * 512 + kb + 8 * half];
            v16bf bf;
#pragma unroll
            for (int i = 0; i < 8; ++i) { bf[i] = pb[i]; bf[8 + i] = pb[16 + i]; }
            oacc[t] = __builtin_amdgcn_wmma_f32_16x16x32_bf16(
                false, a, false, bf, (short)0, oacc[t], false, false);
        }
    }
#pragma unroll
    for (int t = 0; t < 2; ++t)
#pragma unroll
        for (int j = 0; j < 8; ++j) {
            int lr = j + 8 * half;
            out[(m0 + lr) * DMODEL + wave * 32 + t * 16 + row16] = oacc[t][j];
        }
}

// ---------------------------------------------------------------------------
// Host-side launcher
// ---------------------------------------------------------------------------
extern "C" void kernel_launch(void* const* d_in, const int* in_sizes, int n_in,
                              void* d_out, int out_size, void* d_ws, size_t ws_size,
                              hipStream_t stream) {
    const float* x      = (const float*)d_in[0];
    const float* W_in   = (const float*)d_in[1];
    const float* W_out  = (const float*)d_in[2];
    const float* A_real = (const float*)d_in[3];
    const float* A_imag = (const float*)d_in[4];
    const float* B_re   = (const float*)d_in[5];
    const float* B_im   = (const float*)d_in[6];
    const float* C_re   = (const float*)d_in[7];
    const float* C_im   = (const float*)d_in[8];
    const float* Dv     = (const float*)d_in[9];
    const float* inv_dt = (const float*)d_in[10];

    char* ws = (char*)d_ws;
    size_t o_Wt  = 0;
    size_t o_Wo  = o_Wt  + (size_t)1024 * 256 * 2;
    size_t o_dBb = o_Wo  + (size_t)256 * 512 * 2;
    size_t o_Cbm = o_dBb + (size_t)128 * 512 * 2;
    size_t o_dA  = o_Cbm + (size_t)512 * 128 * 2;
    size_t o_u   = o_dA  + 4096;
    size_t o_g   = o_u   + (size_t)MROWS * DINNER * 2;
    size_t o_Bu  = o_g   + (size_t)MROWS * DINNER * 2;
    size_t o_h   = o_Bu  + (size_t)MROWS * 128 * 4;

    __bf16* Wt   = (__bf16*)(ws + o_Wt);
    __bf16* Wo   = (__bf16*)(ws + o_Wo);
    __bf16* dBb  = (__bf16*)(ws + o_dBb);
    __bf16* Cbm  = (__bf16*)(ws + o_Cbm);
    float*  dAc  = (float*)(ws + o_dA);
    __bf16* u_b  = (__bf16*)(ws + o_u);
    __bf16* g_b  = (__bf16*)(ws + o_g);
    float*  Bu   = (float*)(ws + o_Bu);
    float*  hbuf = (float*)(ws + o_h);

    k_setup<<<1024, 256, 0, stream>>>(W_in, W_out, A_real, A_imag, B_re, B_im,
                                      C_re, C_im, inv_dt, Wt, Wo, dBb, Cbm, dAc);
    k_inproj_bu<<<MROWS / 32, 256, 0, stream>>>(x, Wt, dBb, u_b, g_b, Bu);
    k_scan<<<BB * PP, 256, 0, stream>>>(dAc, Bu, hbuf);
    k_out<<<MROWS / 16, 256, 0, stream>>>(hbuf, Cbm, u_b, g_b, Dv, Wo,
                                          (float*)d_out);
}
